// EmbeddingNetwork7_19301583028677
// MI455X (gfx1250) — matrix operand order repaired and checked by reading, not compiled, not run on hardware
//
#include <hip/hip_runtime.h>
#include <hip/hip_bf16.h>
#include <math.h>

typedef _Float16 v16h __attribute__((ext_vector_type(16)));
typedef _Float16 v8h  __attribute__((ext_vector_type(8)));
typedef float    v8f  __attribute__((ext_vector_type(8)));

#define CCH 128          // channels
#define LSEQ 8192        // input length
#define TVAL 8191        // valid timesteps after layer 0 (k=2, no pad)
#define TPAD 8192        // acc stride in time
#define PADT 512         // left zero-pad slots (max dilation tap)
#define PADR 768         // right pad: 512 zero halo + lookahead-stage slack
#define TSTR (PADT + LSEQ + PADR)   // 9472: x buffer time stride
#define NB 32            // batch
#define ROWH 136         // LDS row stride in halves (272 B = 17*16 B, bank-rotating)
// layer weight block (halves): 2 conv * 2 tap * 128 oc * 128 ic
#define WLAYER 65536

// branch-free transcendentals on the v_exp/v_rcp hardware pipes
__device__ __forceinline__ float fsigmoid(float v) {
    return __builtin_amdgcn_rcpf(1.0f + __expf(-v));
}
__device__ __forceinline__ float ftanh(float v) {
    // tanh(x) = 1 - 2/(e^{2x}+1); exact limits at +-inf via rcp
    float e = __expf(2.0f * v);
    return 1.0f - 2.0f * __builtin_amdgcn_rcpf(e + 1.0f);
}

// ---------------------------------------------------------------- std kernel
__global__ void std_kernel(const float* __restrict__ x, float* __restrict__ inv_std) {
    int b = blockIdx.x;
    const float* row = x + (size_t)b * LSEQ;
    float s = 0.f, s2 = 0.f;
    for (int i = threadIdx.x; i < LSEQ; i += 256) {
        float v = row[i];
        s += v; s2 += v * v;
    }
    __shared__ float ls[256], ls2[256];
    ls[threadIdx.x] = s; ls2[threadIdx.x] = s2;
    __syncthreads();
    for (int off = 128; off > 0; off >>= 1) {
        if (threadIdx.x < off) {
            ls[threadIdx.x]  += ls[threadIdx.x + off];
            ls2[threadIdx.x] += ls2[threadIdx.x + off];
        }
        __syncthreads();
    }
    if (threadIdx.x == 0) {
        float sum = ls[0], sum2 = ls2[0];
        float var = (sum2 - sum * sum / (float)LSEQ) / (float)(LSEQ - 1);
        inv_std[b] = 1.0f / (sqrtf(fmaxf(var, 0.0f)) + 1e-8f);
    }
}

// ------------------------------------------- weight prep: f32 -> f16 fragments
__global__ void prep_w(const float* __restrict__ mw, const float* __restrict__ gw,
                       _Float16* __restrict__ wp, int total) {
    int i = blockIdx.x * blockDim.x + threadIdx.x;
    if (i >= total) return;
    int ic    = i & 127;
    int oc    = (i >> 7) & 127;
    int tap   = (i >> 14) & 1;
    int conv  = (i >> 15) & 1;
    int layer = i >> 16;
    const float* src = (conv == 0) ? mw : gw;
    float v = src[(((size_t)(layer * CCH + oc) * CCH + ic) << 1) + tap];
    wp[i] = (_Float16)v;
}

// --------------------------------------- zero the halo pads of both x buffers
__global__ void zeropad_kernel(_Float16* __restrict__ bufA, _Float16* __restrict__ bufB) {
    const int padslots = PADT + (TSTR - (PADT + TVAL));   // 512 + 769 = 1281
    size_t i = (size_t)blockIdx.x * blockDim.x + threadIdx.x;
    size_t total = (size_t)NB * padslots * CCH;
    if (i >= total) return;
    int c = (int)(i & 127);
    size_t s = i >> 7;
    int slot = (int)(s % padslots);
    int b    = (int)(s / padslots);
    int t = (slot < PADT) ? slot : (PADT + TVAL + (slot - PADT));
    size_t idx = ((size_t)b * TSTR + t) * CCH + c;
    bufA[idx] = (_Float16)0;
    bufB[idx] = (_Float16)0;
}

// --------------------------------------------------------------- layer 0
__global__ void layer0_kernel(const float* __restrict__ x, const float* __restrict__ inv_std,
                              const float* __restrict__ mw0, const float* __restrict__ gw0,
                              const float* __restrict__ gb0,
                              _Float16* __restrict__ xo, float* __restrict__ acc) {
    int b = blockIdx.y;
    int g = blockIdx.x * blockDim.x + threadIdx.x;
    int c = g & 127;
    int t = g >> 7;
    if (t >= TVAL) return;
    float inv = inv_std[b];
    float x0 = x[(size_t)b * LSEQ + t]     * inv;
    float x1 = x[(size_t)b * LSEQ + t + 1] * inv;
    float m  = mw0[c * 2] * x0 + mw0[c * 2 + 1] * x1;
    float gg = gw0[c * 2] * x0 + gw0[c * 2 + 1] * x1 + gb0[c];
    float y  = ftanh(m) * fsigmoid(gg);
    xo[((size_t)b * TSTR + PADT + t) * CCH + c] = (_Float16)y;
    acc[((size_t)b * TPAD + t) * CCH + c] = y;
}

// ----------------------------------------------------- dilated gated residual
// A tiles staged once per WG into LDS via async DMA (ASYNCcnt), double-buffered.
// D-tile = 16 timesteps x 16 out-ch; 4 independent WMMA chains; weights in VGPRs.
__global__ __launch_bounds__(256, 1) void wavenet_layer(
    const _Float16* __restrict__ x_in, _Float16* __restrict__ x_out,
    float* __restrict__ acc, const _Float16* __restrict__ wp, int p) {
    // [buf][tap][row 16][136 halves]
    __shared__ _Float16 sA[2][2][16][ROWH];

    const int tid   = threadIdx.x;
    const int b     = blockIdx.y;
    const int chunk = blockIdx.x;            // 32 chunks x 256 timesteps
    const int wave  = tid >> 5;
    const int lane  = tid & 31;
    const int n     = lane & 15;             // tile column (out-channel)
    const int hi    = lane >> 4;             // lane-half select
    const int oc0   = wave * 16;

    const _Float16* xb = x_in + ((size_t)b * TSTR + PADT) * CCH;   // t=0 origin
    _Float16*       xo = x_out + ((size_t)b * TSTR + PADT) * CCH;
    float*          ab = acc + (size_t)b * TPAD * CCH;

    // ---- cooperative async staging assignment (whole WG covers 2 taps x 16 rows)
    const int stap   = tid >> 7;             // which tap region
    const int srow   = (tid >> 3) & 15;      // time row within tile
    const int schk   = (tid & 7) * 2;        // first of 2 consecutive 16B chunks
    const int sshift = stap ? p : -p;        // tap time offset
    const unsigned lds0 = (unsigned)(uintptr_t)&sA[0][stap][srow][schk * 8];
    const unsigned lds1 = (unsigned)(uintptr_t)&sA[1][stap][srow][schk * 8];

    // stage one 16-step tile (both 16B chunks; offset: applies to LDS and global)
    auto stage = [&](int t0n, unsigned ldsOff) {
        const _Float16* g = xb + (size_t)(t0n + srow + sshift) * CCH + schk * 8;
        asm volatile("global_load_async_to_lds_b128 %0, %1, off"
                     :: "v"(ldsOff), "v"(g) : "memory");
        asm volatile("global_load_async_to_lds_b128 %0, %1, off offset:16"
                     :: "v"(ldsOff), "v"(g) : "memory");
    };

    // prime: stage tile 0 into buffer 0 (DMA overlaps weight-register loads)
    stage(chunk * 256, lds0);

    // B fragments (weights): lane n holds column oc0+n, K = kb*32 + hi*16 + [0..15]
    v16h wB[2][2][4];
#pragma unroll
    for (int cv = 0; cv < 2; ++cv)
#pragma unroll
        for (int tp = 0; tp < 2; ++tp)
#pragma unroll
            for (int kb = 0; kb < 4; ++kb) {
                const _Float16* src =
                    wp + ((size_t)((cv * 2 + tp) * CCH + oc0 + n) * CCH) + kb * 32 + hi * 16;
                wB[cv][tp][kb] = *(const v16h*)src;
            }

    const int m  = lane & 15;                // A-matrix row (time within tile)
    const int c0 = hi ? 16 : 0;              // per-lane K chunk start (A layout)

    asm volatile("s_wait_asynccnt 0x0" ::: "memory");
    __syncthreads();

    for (int nt = 0; nt < 16; ++nt) {
        const int t0  = chunk * 256 + nt * 16;
        const int cur = nt & 1;

        // stage next tile into the other LDS buffer while this one computes
        if (nt < 15) stage(t0 + 16, cur ? lds0 : lds1);

        v8f am0 = {}, am1 = {}, ag0 = {}, ag1 = {};
#pragma unroll
        for (int kb = 0; kb < 4; ++kb) {
            const int h0 = kb * 32 + c0;
            v8h l0  = *(const v8h*)&sA[cur][0][m][h0];
            v8h u0  = *(const v8h*)&sA[cur][0][m][h0 + 8];
            v8h l1  = *(const v8h*)&sA[cur][1][m][h0];
            v8h u1  = *(const v8h*)&sA[cur][1][m][h0 + 8];
            v16h a0 = __builtin_shufflevector(l0, u0, 0,1,2,3,4,5,6,7,8,9,10,11,12,13,14,15);
            v16h a1 = __builtin_shufflevector(l1, u1, 0,1,2,3,4,5,6,7,8,9,10,11,12,13,14,15);
            // four independent chains: same-chain WMMA distance = 4
            am0 = __builtin_amdgcn_wmma_f32_16x16x32_f16(false, a0, false, wB[0][0][kb],
                                                         (short)0, am0, false, false);
            ag0 = __builtin_amdgcn_wmma_f32_16x16x32_f16(false, a0, false, wB[1][0][kb],
                                                         (short)0, ag0, false, false);
            am1 = __builtin_amdgcn_wmma_f32_16x16x32_f16(false, a1, false, wB[0][1][kb],
                                                         (short)0, am1, false, false);
            ag1 = __builtin_amdgcn_wmma_f32_16x16x32_f16(false, a1, false, wB[1][1][kb],
                                                         (short)0, ag1, false, false);
        }

        // epilogue: gate, residual, acc update (C/D layout: row M = r + 8*hi).
        // Branchless: only t==8191 is invalid; its x slot must stay zero (halo),
        // its acc slot is writable and never read by the reduction.
        const int tbase = t0 + (hi << 3);
#pragma unroll
        for (int r = 0; r < 8; ++r) {
            int t = tbase + r;
            float y = ftanh(am0[r] + am1[r]) * fsigmoid(ag0[r] + ag1[r]);
            size_t ix = (size_t)t * CCH + oc0 + n;
            float xn = (float)xb[ix] + y;
            xo[ix] = (t < TVAL) ? (_Float16)xn : (_Float16)0;
            ab[ix] = ab[ix] + xn;
        }

        // next tile's DMA must have landed; all waves done reading before reuse
        asm volatile("s_wait_asynccnt 0x0" ::: "memory");
        __syncthreads();
    }
}

// ------------------------------------------------------- max over t + L2 norm
__global__ void maxnorm_kernel(const float* __restrict__ acc, float* __restrict__ out) {
    int b = blockIdx.x;
    int c = threadIdx.x;                     // 128 threads
    const float* base = acc + (size_t)b * TPAD * CCH + c;
    float mx = -3.0e38f;
    for (int t = 0; t < TVAL; ++t) mx = fmaxf(mx, base[(size_t)t * CCH]);
    __shared__ float ss[128];
    ss[c] = mx * mx;
    __syncthreads();
    for (int off = 64; off > 0; off >>= 1) {
        if (c < off) ss[c] += ss[c + off];
        __syncthreads();
    }
    float norm = sqrtf(ss[0]);
    out[b * CCH + c] = mx / (norm + 1e-8f);
}

// ------------------------------------------------------------------- launcher
extern "C" void kernel_launch(void* const* d_in, const int* in_sizes, int n_in,
                              void* d_out, int out_size, void* d_ws, size_t ws_size,
                              hipStream_t stream) {
    const float* x   = (const float*)d_in[0];   // (32, 8192)
    const float* mw0 = (const float*)d_in[1];   // (128, 1, 2)
    const float* gw0 = (const float*)d_in[2];   // (128, 1, 2)
    const float* gb0 = (const float*)d_in[3];   // (128,)
    const float* mw  = (const float*)d_in[4];   // (10, 128, 128, 2)
    const float* gw  = (const float*)d_in[5];   // (10, 128, 128, 2)
    float* out = (float*)d_out;                 // (32, 128)

    char* ws = (char*)d_ws;
    float*    inv_std = (float*)ws;                           // 256 B slot
    _Float16* wp      = (_Float16*)(ws + 256);                // 10*65536 halves
    size_t off = 256 + (size_t)10 * WLAYER * sizeof(_Float16);
    _Float16* bufA = (_Float16*)(ws + off); off += (size_t)NB * TSTR * CCH * sizeof(_Float16);
    _Float16* bufB = (_Float16*)(ws + off); off += (size_t)NB * TSTR * CCH * sizeof(_Float16);
    float*    acc  = (float*)(ws + off);

    std_kernel<<<NB, 256, 0, stream>>>(x, inv_std);
    prep_w<<<(10 * WLAYER + 255) / 256, 256, 0, stream>>>(mw, gw, wp, 10 * WLAYER);
    {
        const int padslots = PADT + (TSTR - (PADT + TVAL));
        size_t total = (size_t)NB * padslots * CCH;
        zeropad_kernel<<<(unsigned)((total + 255) / 256), 256, 0, stream>>>(bufA, bufB);
    }
    layer0_kernel<<<dim3(4096, NB), 256, 0, stream>>>(x, inv_std, mw0, gw0, gb0, bufA, acc);

    _Float16* xin = bufA;
    _Float16* xout = bufB;
    for (int i = 0; i < 10; ++i) {           // dilation 2^(i+1), taps at t +/- 2^i
        wavenet_layer<<<dim3(32, NB), 256, 0, stream>>>(xin, xout, acc,
                                                        wp + (size_t)i * WLAYER, 1 << i);
        _Float16* tmp = xin; xin = xout; xout = tmp;
    }
    maxnorm_kernel<<<NB, 128, 0, stream>>>(acc, out);
}
